// matcher_41918880809178
// MI455X (gfx1250) — matrix operand order repaired and checked by reading, not compiled, not run on hardware
//
#include <hip/hip_runtime.h>
#include <hip/hip_bf16.h>
#include <cstring>

// ---------------------------------------------------------------------------
// CDNA5 wave32 WMMA types
// ---------------------------------------------------------------------------
typedef __attribute__((ext_vector_type(16))) _Float16 v16h;
typedef __attribute__((ext_vector_type(8)))  float    v8f;

union AFrag { v16h v; _Float16 e[16]; };
union CFrag { v8f  v; float     e[8]; };

// Async global->LDS (ASYNCcnt) support, probe-guarded with sync fallback
#if defined(__has_builtin)
#  if __has_builtin(__builtin_amdgcn_global_load_async_to_lds_b128) && \
      __has_builtin(__builtin_amdgcn_s_wait_asynccnt)
#    define USE_ASYNC_LDS 1
#  endif
#endif
#ifndef USE_ASYNC_LDS
#  define USE_ASYNC_LDS 0
#endif

#if USE_ASYNC_LDS
typedef int v4i __attribute__((vector_size(16)));
typedef __attribute__((address_space(1))) v4i gv4i;   // global (AS1)
typedef __attribute__((address_space(3))) v4i lv4i;   // LDS (AS3)
#endif

// ---------------------------------------------------------------------------
// Problem constants (match reference)
// ---------------------------------------------------------------------------
#define BB   4
#define NN   2048
#define CC   128
#define HH   4
#define TK   256        // TOPK
#define C2   256        // 2*C
#define DIM  2049       // N+1 (sinkhorn)
#define BN_INV 0.9999950000374997f   // 1/sqrt(1+1e-5)

// GEMM epilogue flags
#define GF_BIAS   1
#define GF_BN     2
#define GF_RELU   4
#define GF_RESID  8
#define GF_WF     16
#define GF_WH     32
#define GF_ACCUM  64
#define GF_BT     128   // B stored K x N (NN-form); default is N x K (NT-form)
#define GF_DMSQRT 256   // v = sqrt(|2-2v|)

struct GemmP {
  const _Float16* A; const _Float16* Bm;
  float* OF; _Float16* OH;
  const float* bias; const float* bnG; const float* bnB; const float* R;
  long sA1, sA2, sB1, sB2, sO1, sO2, sR1, sR2;
  int lda, ldb, ldo, ldr;
  int M, N, K, zdiv, flags;
  float alpha;
};

// 32B (16-half) global -> LDS copy; async on CDNA5 when available
__device__ __forceinline__ void copy16h(const _Float16* g, _Float16* l) {
#if USE_ASYNC_LDS
  __builtin_amdgcn_global_load_async_to_lds_b128((gv4i*)g, (lv4i*)l, 0, 0);
  __builtin_amdgcn_global_load_async_to_lds_b128((gv4i*)g, (lv4i*)l, 16, 0);
#else
  const float4* s = (const float4*)g;
  float4 a = s[0], b = s[1];
  ((float4*)l)[0] = a; ((float4*)l)[1] = b;
#endif
}
__device__ __forceinline__ void pipe_sync() {
#if USE_ASYNC_LDS
  __builtin_amdgcn_s_wait_asynccnt(0);
#endif
  __syncthreads();
}

// ---------------------------------------------------------------------------
// Batched WMMA GEMM:  Out[z] = alpha * A(MxK) * B^T(NxK)  (+ fused epilogue)
// 128 threads = 4 wave32; block tile 64x64, K-step 32.
// Double-buffered LDS pipeline: async global->LDS loads of the next K-tile are
// issued before computing the current one; per-step sync is s_wait_asynccnt 0
// + one workgroup barrier.  OOB rows are simply not loaded: stale LDS there
// only feeds accumulator rows/cols that the epilogue discards.
// ---------------------------------------------------------------------------
__global__ __launch_bounds__(128) void k_gemm(GemmP p) {
  __shared__ __align__(16) _Float16 As[2][64][40];
  __shared__ __align__(16) _Float16 Bs[2][64][40];
  const int z  = blockIdx.z;
  const int zq = z / p.zdiv, zr = z % p.zdiv;
  const _Float16* A  = p.A  + zq * p.sA1 + zr * p.sA2;
  const _Float16* Bm = p.Bm + zq * p.sB1 + zr * p.sB2;
  const long oo  = zq * p.sO1 + zr * p.sO2;
  const long orr = zq * p.sR1 + zr * p.sR2;
  const int tileM = blockIdx.y * 64, tileN = blockIdx.x * 64;
  const int tid = threadIdx.x, wave = tid >> 5, lane = tid & 31;

  // loader thread mappings
  const int rL  = tid >> 1, kbL = (tid & 1) * 16;   // A and NT-B: 64 rows x 32 halves
  const int grA = tileM + rL;
  const int gjB = tileN + rL;
  const int kBT = tid >> 2, jcBT = (tid & 3) * 16;  // BT-B: transpose scatter

  auto load_tiles = [&](int kk, int buf) {
    if (grA < p.M)
      copy16h(A + (long)grA * p.lda + kk + kbL, &As[buf][rL][kbL]);
    if (!(p.flags & GF_BT)) {
      if (gjB < p.N)
        copy16h(Bm + (long)gjB * p.ldb + kk + kbL, &Bs[buf][rL][kbL]);
    } else {
      int gk = kk + kBT;
      __align__(16) _Float16 tmp[16];
      if (tileN + jcBT + 16 <= p.N) {
        const float4* bp = (const float4*)(Bm + (long)gk * p.ldb + tileN + jcBT);
        *(float4*)&tmp[0] = bp[0]; *(float4*)&tmp[8] = bp[1];
      } else {
#pragma unroll
        for (int t = 0; t < 16; ++t)
          tmp[t] = (tileN + jcBT + t < p.N)
                       ? Bm[(long)gk * p.ldb + tileN + jcBT + t] : (_Float16)0.f;
      }
#pragma unroll
      for (int t = 0; t < 16; ++t) Bs[buf][jcBT + t][kBT] = tmp[t];
    }
  };

  CFrag acc[4];
#pragma unroll
  for (int c = 0; c < 4; ++c)
#pragma unroll
    for (int i = 0; i < 8; ++i) acc[c].e[i] = 0.f;

  int buf = 0;
  load_tiles(0, 0);
  pipe_sync();

  for (int kk = 0; kk < p.K; kk += 32) {
    if (kk + 32 < p.K) load_tiles(kk + 32, buf ^ 1);   // prefetch next K-tile

    // ---- fragments per ISA VGPR layout, 4 WMMAs per wave per K-step
    AFrag af;
    const int mr  = wave * 16 + (lane & 15);
    const int koA = (lane >= 16) ? 8  : 0;
    const int koB = (lane >= 16) ? 16 : 0;
#pragma unroll
    for (int h = 0; h < 16; ++h) {
      int kl = h + ((h >= 8) ? 8 : 0) + koA;
      af.e[h] = As[buf][mr][kl];
    }
#pragma unroll
    for (int c = 0; c < 4; ++c) {
      AFrag bf;
      int col = c * 16 + (lane & 15);
#pragma unroll
      for (int h = 0; h < 16; ++h) bf.e[h] = Bs[buf][col][h + ((h >= 8) ? 8 : 0) + koA];
      acc[c].v = __builtin_amdgcn_wmma_f32_16x16x32_f16(
          false, af.v, false, bf.v, (short)0, acc[c].v, false, false);
    }
    pipe_sync();
    buf ^= 1;
  }

  // ---- fused epilogue
#pragma unroll
  for (int c = 0; c < 4; ++c) {
#pragma unroll
    for (int vi = 0; vi < 8; ++vi) {
      int r   = tileM + wave * 16 + vi + ((lane >= 16) ? 8 : 0);
      int col = tileN + c * 16 + (lane & 15);
      if (r < p.M && col < p.N) {
        float v = acc[c].e[vi] * p.alpha;
        long idx = oo + (long)r * p.ldo + col;
        if (p.flags & GF_ACCUM)  v += p.OF[idx];
        if (p.flags & GF_BIAS)   v += p.bias[col];
        if (p.flags & GF_BN)     v = v * (p.bnG[col] * BN_INV) + p.bnB[col];
        if (p.flags & GF_RELU)   v = fmaxf(v, 0.f);
        if (p.flags & GF_RESID)  v += p.R[orr + (long)r * p.ldr + col];
        if (p.flags & GF_DMSQRT) v = sqrtf(fabsf(2.f - 2.f * v));
        if (p.flags & (GF_WF | GF_ACCUM)) p.OF[idx] = v;
        if (p.flags & GF_WH)              p.OH[idx] = (_Float16)v;
      }
    }
  }
}

// ---------------------------------------------------------------------------
// Small utility / elementwise kernels
// ---------------------------------------------------------------------------
__global__ void k_h(const float* s, _Float16* d, int n) {
  int i = blockIdx.x * 256 + threadIdx.x;
  if (i < n) d[i] = (_Float16)s[i];
}
__global__ void k_fill(float* p, float v, int n) {
  int i = blockIdx.x * 256 + threadIdx.x;
  if (i < n) p[i] = v;
}

__global__ void k_l2norm(const float* d, float* of, _Float16* oh) {
  int pnt = blockIdx.x, c = threadIdx.x;                 // 128 thr
  float v = d[(long)pnt * CC + c];
  __shared__ float red[128];
  red[c] = v * v; __syncthreads();
  for (int s = 64; s > 0; s >>= 1) { if (c < s) red[c] += red[c + s]; __syncthreads(); }
  float o = v / fmaxf(sqrtf(red[0]), 1e-12f);
  of[(long)pnt * CC + c] = o;
  oh[(long)pnt * CC + c] = (_Float16)o;
}

// row-wise softmax; optional per-column weight wv; f32 and/or f16 output; safe in place
__global__ void k_softmax(const float* S, float* OF, _Float16* OH, const float* wv,
                          int m, long zstride, int hdiv, int wvm) {
  int z = blockIdx.y, row = blockIdx.x, tid = threadIdx.x;
  const float* src = S + z * zstride + (long)row * m;
  __shared__ float red[256];
  float mx = -1e30f;
  for (int j = tid; j < m; j += 256) mx = fmaxf(mx, src[j]);
  red[tid] = mx; __syncthreads();
  for (int s = 128; s > 0; s >>= 1) { if (tid < s) red[tid] = fmaxf(red[tid], red[tid+s]); __syncthreads(); }
  mx = red[0]; __syncthreads();
  float sum = 0.f;
  for (int j = tid; j < m; j += 256) sum += __expf(src[j] - mx);
  red[tid] = sum; __syncthreads();
  for (int s = 128; s > 0; s >>= 1) { if (tid < s) red[tid] += red[tid+s]; __syncthreads(); }
  float inv = 1.f / red[0];
  long ob = z * zstride + (long)row * m;
  for (int j = tid; j < m; j += 256) {
    float pv = __expf(src[j] - mx) * inv;
    if (wv) pv *= wv[(z / hdiv) * wvm + j];
    if (OF) OF[ob + j] = pv;
    if (OH) OH[ob + j] = (_Float16)pv;
  }
}

// InstanceNorm over points (per channel) + BN + ReLU -> f16
__global__ void k_inorm(const float* src, int ld, int npts, const float* g,
                        const float* bt, _Float16* dst, int dld) {
  int b = blockIdx.y, c = blockIdx.x, tid = threadIdx.x;
  const float* S = src + (long)b * npts * ld + c;
  float s1 = 0.f, s2 = 0.f;
  for (int i = tid; i < npts; i += 256) { float v = S[(long)i * ld]; s1 += v; s2 += v * v; }
  __shared__ float r1[256], r2[256];
  r1[tid] = s1; r2[tid] = s2; __syncthreads();
  for (int s = 128; s > 0; s >>= 1) {
    if (tid < s) { r1[tid] += r1[tid+s]; r2[tid] += r2[tid+s]; } __syncthreads();
  }
  float mean = r1[0] / npts;
  float var  = r2[0] / npts - mean * mean;
  float inv  = rsqrtf(fmaxf(var, 0.f) + 1e-3f);
  float gg = g[c] * BN_INV, bb = bt[c];
  _Float16* D = dst + (long)b * npts * dld + c;
  for (int i = tid; i < npts; i += 256) {
    float v = (S[(long)i * ld] - mean) * inv * gg + bb;
    D[(long)i * dld] = (_Float16)fmaxf(v, 0.f);
  }
}

// pos-enc layer 0 (Cin=2) + bn + relu -> f16
__global__ void k_pe0(const float* x, const float* W, const float* bias,
                      const float* g, const float* bt, _Float16* outH, int total) {
  int gid = blockIdx.x * 256 + threadIdx.x;
  int p = gid >> 5, o = gid & 31;
  if (p >= total) return;
  float v = W[o*2] * x[p*2] + W[o*2+1] * x[p*2+1] + bias[o];
  v = v * (g[o] * BN_INV) + bt[o];
  outH[(long)p * 32 + o] = (_Float16)fmaxf(v, 0.f);
}

// conf head: sigmoid(c2.h2 + sc.x), o=1 channel
__global__ void k_conf(const _Float16* h2, const float* c2W, const float* c2b,
                       const float* scW, const float* scb,
                       const float* c1f, const float* c2f, float* sw, int npts, int total) {
  int gid = blockIdx.x * 64 + threadIdx.x;
  if (gid >= total) return;
  int b = gid / npts, p = gid % npts;
  float acc = c2b[0] + scb[0];
  const _Float16* H  = h2  + ((long)b * npts + p) * C2;
  const float*    X1 = c1f + ((long)b * npts + p) * CC;
  const float*    X2 = c2f + ((long)b * npts + p) * CC;
  for (int c = 0; c < C2; ++c) acc += c2W[c] * (float)H[c];
  for (int c = 0; c < CC; ++c) acc += scW[c] * X1[c] + scW[c + CC] * X2[c];
  sw[gid] = 1.f / (1.f + __expf(-acc));
}

__global__ void k_gather(const float* src, const _Float16* srch, const int* idx,
                         float* dst, _Float16* dsth, int Npts, int Kc) {
  int b = blockIdx.y, k = blockIdx.x, c = threadIdx.x;    // 128 thr
  int id = idx[b * Kc + k];
  float v = src[((long)b * Npts + id) * CC + c];
  (void)srch;
  dst [((long)b * Kc + k) * CC + c] = v;
  dsth[((long)b * Kc + k) * CC + c] = (_Float16)v;
}

__global__ void k_gxy(const float* x2, const int* nn1, float* x2g, int Npts, int total) {
  int gid = blockIdx.x * 256 + threadIdx.x;
  if (gid >= total) return;
  int b = gid / Npts, j = nn1[gid];
  x2g[gid*2]   = x2[((long)b*Npts + j)*2];
  x2g[gid*2+1] = x2[((long)b*Npts + j)*2+1];
}

// ---- seeding -------------------------------------------------------------
__global__ void k_row_top2(const _Float16* dm, float* v1, float* v2, int* nn1, int Npts) {
  int row = blockIdx.x, tid = threadIdx.x;
  const _Float16* p = dm + (long)row * Npts;
  float m1 = 1e30f, m2 = 1e30f; int i1 = 0;
  for (int j = tid; j < Npts; j += 256) {
    float d = (float)p[j];
    if (d < m1) { m2 = m1; m1 = d; i1 = j; } else if (d < m2) m2 = d;
  }
  __shared__ float a1[256], a2[256]; __shared__ int ai[256];
  a1[tid] = m1; a2[tid] = m2; ai[tid] = i1; __syncthreads();
  for (int s = 128; s > 0; s >>= 1) {
    if (tid < s) {
      float b1 = a1[tid+s], b2 = a2[tid+s]; int bi = ai[tid+s];
      if (b1 < a1[tid]) { a2[tid] = fminf(a1[tid], b2); a1[tid] = b1; ai[tid] = bi; }
      else              { a2[tid] = fminf(a2[tid], b1); }
    }
    __syncthreads();
  }
  if (tid == 0) { v1[row] = a1[0]; v2[row] = a2[0]; nn1[row] = ai[0]; }
}

__global__ void k_colmin(const _Float16* dm, int* nn2, int Npts) {
  int b = blockIdx.y, j = blockIdx.x * 256 + threadIdx.x;
  if (j >= Npts) return;
  const _Float16* D = dm + (long)b * Npts * Npts + j;
  float best = 1e30f; int bi = 0;
  for (int i = 0; i < Npts; ++i) { float v = (float)D[(long)i * Npts]; if (v < best) { best = v; bi = i; } }
  nn2[b * Npts + j] = bi;
}

__global__ void k_score(const float* v1, const float* v2, const int* nn1, const int* nn2,
                        float* sc, int Npts, int total) {
  int gid = blockIdx.x * 256 + threadIdx.x;
  if (gid >= total) return;
  int b = gid / Npts, i = gid % Npts;
  float s = v2[gid] / fmaxf(v1[gid], 1e-12f);
  if (nn2[b * Npts + nn1[gid]] != i) s = -1.f;     // mutual-NN check
  sc[gid] = s;
}

__global__ void k_pdist_sum(const float* x, float* sums, int Npts) {
  int b = blockIdx.y, i = blockIdx.x * 256 + threadIdx.x;
  const float* X = x + (long)b * Npts * 2;
  float acc = 0.f;
  if (i < Npts) {
    float xi = X[2*i], yi = X[2*i+1];
    for (int j = 0; j < Npts; ++j) {
      float dx = xi - X[2*j], dy = yi - X[2*j+1];
      acc += sqrtf(dx*dx + dy*dy);
    }
  }
  __shared__ float red[256];
  red[threadIdx.x] = acc; __syncthreads();
  for (int s = 128; s > 0; s >>= 1) { if (threadIdx.x < s) red[threadIdx.x] += red[threadIdx.x+s]; __syncthreads(); }
  if (threadIdx.x == 0) atomicAdd(&sums[b], red[0]);
}

// NMS local-max with on-the-fly distances (avoids 2x N^2 matrices)
__global__ void k_localmax(const float* sIn, float* sOut, const float* x1, const float* x2g,
                           const float* sum1, const float* sum2, int Npts) {
  int b = blockIdx.y, i = blockIdx.x;
  float r1 = 0.01f * sum1[b] / ((float)Npts * (float)Npts);
  float r2 = 0.01f * sum2[b] / ((float)Npts * (float)Npts);
  const float* X1 = x1  + (long)b * Npts * 2;
  const float* X2 = x2g + (long)b * Npts * 2;
  float xi = X1[2*i], yi = X1[2*i+1], ui = X2[2*i], vi = X2[2*i+1];
  float si = sIn[b * Npts + i];
  int ok = 1;
  for (int j = threadIdx.x; j < Npts; j += 256) {
    float sj = sIn[b * Npts + j];
    if (si < sj) {
      float dx = xi - X1[2*j], dy = yi - X1[2*j+1];
      float du = ui - X2[2*j], dv = vi - X2[2*j+1];
      if (!(sqrtf(dx*dx + dy*dy) >= r1 && sqrtf(du*du + dv*dv) >= r2)) ok = 0;
    }
  }
  __shared__ int red[256];
  red[threadIdx.x] = ok; __syncthreads();
  for (int s = 128; s > 0; s >>= 1) { if (threadIdx.x < s) red[threadIdx.x] &= red[threadIdx.x+s]; __syncthreads(); }
  if (threadIdx.x == 0)
    sOut[b * Npts + i] = (red[0] && si >= 1.11f) ? si : -1.f;
}

__global__ void k_topk(const float* score, const int* nn1, int* s1, int* s2, int Npts, int K) {
  int b = blockIdx.x, tid = threadIdx.x;                 // 256 thr
  __shared__ float sc[NN];
  __shared__ float rv[256]; __shared__ int ri[256];
  for (int i = tid; i < Npts; i += 256) sc[i] = score[b * Npts + i];
  __syncthreads();
  for (int k = 0; k < K; ++k) {
    float best = -1e30f; int bi = 0;
    for (int i = tid; i < Npts; i += 256) if (sc[i] > best) { best = sc[i]; bi = i; }
    rv[tid] = best; ri[tid] = bi; __syncthreads();
    for (int s = 128; s > 0; s >>= 1) {
      if (tid < s && rv[tid+s] > rv[tid]) { rv[tid] = rv[tid+s]; ri[tid] = ri[tid+s]; }
      __syncthreads();
    }
    if (tid == 0) {
      int idx = ri[0];
      s1[b*K + k] = idx;
      s2[b*K + k] = nn1[b*Npts + idx];
      sc[idx] = -1e30f;
    }
    __syncthreads();
  }
}

// ---- sinkhorn ------------------------------------------------------------
__global__ void k_border(float* P, const float* dustbin, int n) {
  int b = blockIdx.y;
  long base = (long)b * (n+1) * (n+1);
  float db = *dustbin;
  int i = blockIdx.x * 256 + threadIdx.x;
  if (i <= n) {
    P[base + (long)n * (n+1) + i] = db;
    P[base + (long)i * (n+1) + n] = db;
  }
}
__global__ void k_sink_u(const float* P, const float* v, float* u, int n) {
  int dim = n + 1, b = blockIdx.y, i = blockIdx.x, tid = threadIdx.x;
  const float* row = P + ((long)b * dim + i) * dim;
  const float* vb  = v + b * dim;
  float acc = 0.f;
  for (int j = tid; j < dim; j += 256) acc += row[j] * vb[j];
  __shared__ float red[256];
  red[tid] = acc; __syncthreads();
  for (int s = 128; s > 0; s >>= 1) { if (tid < s) red[tid] += red[tid+s]; __syncthreads(); }
  if (tid == 0) u[b*dim + i] = ((i == n) ? (float)(n+1) : 1.f) / (red[0] + 1e-8f);
}
__global__ void k_sink_v(const float* P, const float* u, float* v, int n) {
  int dim = n + 1, b = blockIdx.y, j = blockIdx.x * 256 + threadIdx.x;
  if (j >= dim) return;
  const float* ub = u + b * dim;
  float acc = 0.f;
  for (int i = 0; i < dim; ++i) acc += P[((long)b*dim + i)*dim + j] * ub[i];
  v[b*dim + j] = ((j == n) ? (float)(n+1) : 1.f) / (acc + 1e-8f);
}
__global__ void k_sink_scale(float* P, const float* u, const float* v, int n) {
  int dim = n + 1, b = blockIdx.y, i = blockIdx.x;
  float ui = u[b*dim + i];
  float* row = P + ((long)b * dim + i) * dim;
  const float* vb = v + b * dim;
  for (int j = threadIdx.x; j < dim; j += 256) row[j] *= ui * vb[j];
}

// ---------------------------------------------------------------------------
// Host orchestration
// ---------------------------------------------------------------------------
struct Arena {
  char* base; size_t off;
  void* alloc(size_t bytes) {
    size_t o = (off + 255) & ~(size_t)255;
    off = o + bytes;
    return base + o;
  }
};
struct Cur { void* const* d_in; int idx; };
static const float* take(Cur& c) { return (const float*)c.d_in[c.idx++]; }

struct ConvP { const float* W; const float* b; const _Float16* Wh; };
struct BnP   { const float* g; const float* bt; };
struct AttnW { ConvP q, k, v, mh, cat1, cat2; BnP cat_bn; };
struct PcnW  { BnP bn1, bn2; ConvP c1, c2, sc; };
struct HybW  { ConvP cf1, cf2; BnP cf_bn; PcnW conf; AttnW cross, down, self_, up; };
struct PeW   { BnP bn0, bn1, bn2, bn3; ConvP c0, c1, c2, c3, c4; };

static const _Float16* conv16(Arena& ar, hipStream_t st, const float* W, int n) {
  _Float16* d = (_Float16*)ar.alloc((size_t)n * 2);
  k_h<<<(n + 255) / 256, 256, 0, st>>>(W, d, n);
  return d;
}
static ConvP rdConv(Cur& c, Arena& ar, hipStream_t st, int o, int i) {
  ConvP w; w.W = take(c); w.b = take(c); w.Wh = conv16(ar, st, w.W, o * i); return w;
}
static BnP rdBn(Cur& c) { BnP b; b.bt = take(c); b.g = take(c); return b; }
static AttnW rdAttn(Cur& c, Arena& ar, hipStream_t st) {   // alphabetical leaf order
  AttnW a;
  a.cat1 = rdConv(c, ar, st, C2, C2);
  a.cat2 = rdConv(c, ar, st, CC, C2);
  a.cat_bn = rdBn(c);
  a.k  = rdConv(c, ar, st, CC, CC);
  a.mh = rdConv(c, ar, st, CC, CC);
  a.q  = rdConv(c, ar, st, CC, CC);
  a.v  = rdConv(c, ar, st, CC, CC);
  return a;
}
static PcnW rdPcn(Cur& c, Arena& ar, hipStream_t st) {
  PcnW p;
  p.bn1 = rdBn(c); p.bn2 = rdBn(c);
  p.c1 = rdConv(c, ar, st, C2, C2);
  p.c2 = rdConv(c, ar, st, 1, C2);
  p.sc = rdConv(c, ar, st, 1, C2);
  return p;
}
static HybW rdHyb(Cur& c, Arena& ar, hipStream_t st) {
  HybW h;
  h.cf1 = rdConv(c, ar, st, C2, C2);
  h.cf2 = rdConv(c, ar, st, C2, C2);
  h.cf_bn = rdBn(c);
  h.conf  = rdPcn(c, ar, st);
  h.cross = rdAttn(c, ar, st);
  h.down  = rdAttn(c, ar, st);
  h.self_ = rdAttn(c, ar, st);
  h.up    = rdAttn(c, ar, st);
  return h;
}

struct Act { float* f; _Float16* h; int n; int ld; };
struct Scratch { _Float16 *tQ, *tK, *tV, *tAdd, *tMh, *tP, *h1h; float *tS, *h1; };

static GemmP gp0() { GemmP p; memset(&p, 0, sizeof(p)); p.zdiv = 1; p.alpha = 1.f; return p; }
static void gemm_launch(hipStream_t st, const GemmP& p, int Z) {
  dim3 g((p.N + 63) / 64, (p.M + 63) / 64, Z);
  k_gemm<<<g, 128, 0, st>>>(p);
}

static void attn_apply(hipStream_t st, const AttnW& w, Act in1, Act in2,
                       const float* wv, Act out, Scratch& s) {
  const int n = in1.n, m = in2.n;
  { GemmP p = gp0();                                           // q
    p.A = in1.h; p.sA1 = (long)n * in1.ld; p.lda = in1.ld;
    p.Bm = w.q.Wh; p.ldb = CC;
    p.OH = s.tQ; p.sO1 = (long)n * CC; p.ldo = CC;
    p.bias = w.q.b; p.M = n; p.N = CC; p.K = CC;
    p.flags = GF_BIAS | GF_WH; gemm_launch(st, p, BB); }
  { GemmP p = gp0();                                           // k
    p.A = in2.h; p.sA1 = (long)m * in2.ld; p.lda = in2.ld;
    p.Bm = w.k.Wh; p.ldb = CC;
    p.OH = s.tK; p.sO1 = (long)m * CC; p.ldo = CC;
    p.bias = w.k.b; p.M = m; p.N = CC; p.K = CC;
    p.flags = GF_BIAS | GF_WH; gemm_launch(st, p, BB); }
  { GemmP p = gp0();                                           // v
    p.A = in2.h; p.sA1 = (long)m * in2.ld; p.lda = in2.ld;
    p.Bm = w.v.Wh; p.ldb = CC;
    p.OH = s.tV; p.sO1 = (long)m * CC; p.ldo = CC;
    p.bias = w.v.b; p.M = m; p.N = CC; p.K = CC;
    p.flags = GF_BIAS | GF_WH; gemm_launch(st, p, BB); }
  { GemmP p = gp0(); p.zdiv = HH; p.alpha = 0.17677669529663687f;   // Q K^T / sqrt(hd)
    p.A = s.tQ; p.sA1 = (long)n * CC; p.sA2 = 32; p.lda = CC;
    p.Bm = s.tK; p.sB1 = (long)m * CC; p.sB2 = 32; p.ldb = CC;
    p.OF = s.tS; p.sO1 = (long)HH * n * m; p.sO2 = (long)n * m; p.ldo = m;
    p.M = n; p.N = m; p.K = 32; p.flags = GF_WF; gemm_launch(st, p, BB * HH); }
  k_softmax<<<dim3(n, BB * HH), 256, 0, st>>>(s.tS, nullptr, s.tP, wv,
                                              m, (long)n * m, HH, m);
  { GemmP p = gp0(); p.zdiv = HH;                               // P V
    p.A = s.tP; p.sA1 = (long)HH * n * m; p.sA2 = (long)n * m; p.lda = m;
    p.Bm = s.tV; p.sB1 = (long)m * CC; p.sB2 = 32; p.ldb = CC;
    p.OH = s.tAdd; p.sO1 = (long)n * CC; p.sO2 = 32; p.ldo = CC;
    p.M = n; p.N = 32; p.K = m; p.flags = GF_WH | GF_BT; gemm_launch(st, p, BB * HH); }
  { GemmP p = gp0();                                            // mh
    p.A = s.tAdd; p.sA1 = (long)n * CC; p.lda = CC;
    p.Bm = w.mh.Wh; p.ldb = CC;
    p.OH = s.tMh; p.sO1 = (long)n * CC; p.ldo = CC;
    p.bias = w.mh.b; p.M = n; p.N = CC; p.K = CC;
    p.flags = GF_BIAS | GF_WH; gemm_launch(st, p, BB); }
  { GemmP p = gp0();                                            // cat1 part A (d1)
    p.A = in1.h; p.sA1 = (long)n * in1.ld; p.lda = in1.ld;
    p.Bm = w.cat1.Wh; p.ldb = C2;
    p.OF = s.h1; p.sO1 = (long)n * C2; p.ldo = C2;
    p.M = n; p.N = C2; p.K = CC; p.flags = GF_WF; gemm_launch(st, p, BB); }
  { GemmP p = gp0();                                            // cat1 part B (+bias,bn,relu)
    p.A = s.tMh; p.sA1 = (long)n * CC; p.lda = CC;
    p.Bm = w.cat1.Wh + CC; p.ldb = C2;
    p.OF = s.h1; p.OH = s.h1h; p.sO1 = (long)n * C2; p.ldo = C2;
    p.bias = w.cat1.b; p.bnG = w.cat_bn.g; p.bnB = w.cat_bn.bt;
    p.M = n; p.N = C2; p.K = CC;
    p.flags = GF_ACCUM | GF_BIAS | GF_BN | GF_RELU | GF_WH; gemm_launch(st, p, BB); }
  { GemmP p = gp0();                                            // cat2 + residual
    p.A = s.h1h; p.sA1 = (long)n * C2; p.lda = C2;
    p.Bm = w.cat2.Wh; p.ldb = C2;
    p.OF = out.f; p.OH = out.h; p.sO1 = (long)n * out.ld; p.ldo = out.ld;
    p.bias = w.cat2.b;
    p.R = in1.f; p.sR1 = (long)n * in1.ld; p.ldr = in1.ld;
    p.M = n; p.N = CC; p.K = C2;
    p.flags = GF_BIAS | GF_RESID | GF_WF | GF_WH; gemm_launch(st, p, BB); }
}

extern "C" void kernel_launch(void* const* d_in, const int* in_sizes, int n_in,
                              void* d_out, int out_size, void* d_ws, size_t ws_size,
                              hipStream_t stream) {
  (void)in_sizes; (void)n_in; (void)out_size; (void)ws_size;
  const float* x1    = (const float*)d_in[0];
  const float* x2    = (const float*)d_in[1];
  const float* desc1 = (const float*)d_in[2];
  const float* desc2 = (const float*)d_in[3];

  Arena ar{(char*)d_ws, 0};
  Cur   cur{d_in, 4};

  // -------- parse params (alphabetical pytree order) + f16 weight shadows
  HybW blocks[6];
  for (int i = 0; i < 6; ++i) blocks[i] = rdHyb(cur, ar, stream);
  const float* dustbin = take(cur);
  ConvP finalW = rdConv(cur, ar, stream, CC, CC);
  PeW pe;
  pe.bn0 = rdBn(cur); pe.bn1 = rdBn(cur); pe.bn2 = rdBn(cur); pe.bn3 = rdBn(cur);
  pe.c0 = rdConv(cur, ar, stream, 32, 2);
  pe.c1 = rdConv(cur, ar, stream, 64, 32);
  pe.c2 = rdConv(cur, ar, stream, 128, 64);
  pe.c3 = rdConv(cur, ar, stream, 256, 128);
  pe.c4 = rdConv(cur, ar, stream, CC, 256);

  // -------- workspace
  const long PN = (long)BB * NN;
  float*    d1n = (float*)ar.alloc(PN * CC * 4);
  float*    d2n = (float*)ar.alloc(PN * CC * 4);
  _Float16* d1h = (_Float16*)ar.alloc(PN * CC * 2);
  _Float16* d2h = (_Float16*)ar.alloc(PN * CC * 2);
  float*    a1f = (float*)ar.alloc(PN * CC * 4);
  float*    a2f = (float*)ar.alloc(PN * CC * 4);
  _Float16* a1h = (_Float16*)ar.alloc(PN * CC * 2);
  _Float16* a2h = (_Float16*)ar.alloc(PN * CC * 2);
  _Float16* dm  = (_Float16*)ar.alloc((long)BB * NN * NN * 2);
  float* v1  = (float*)ar.alloc(PN * 4);
  float* v2  = (float*)ar.alloc(PN * 4);
  float* sc0 = (float*)ar.alloc(PN * 4);
  float* sc1 = (float*)ar.alloc(PN * 4);
  int* nn1 = (int*)ar.alloc(PN * 4);
  int* nn2 = (int*)ar.alloc(PN * 4);
  float* x2g  = (float*)ar.alloc(PN * 2 * 4);
  float* sums = (float*)ar.alloc(8 * 4);
  int* s1 = (int*)ar.alloc((long)BB * TK * 4);
  int* s2 = (int*)ar.alloc((long)BB * TK * 4);
  _Float16* t0h = (_Float16*)ar.alloc(PN * 32 * 2);
  _Float16* t1h = (_Float16*)ar.alloc(PN * 64 * 2);
  _Float16* t2h = (_Float16*)ar.alloc(PN * 128 * 2);
  _Float16* t3h = (_Float16*)ar.alloc(PN * 256 * 2);
  Scratch s;
  s.tQ   = (_Float16*)ar.alloc(PN * CC * 2);
  s.tK   = (_Float16*)ar.alloc(PN * CC * 2);
  s.tV   = (_Float16*)ar.alloc(PN * CC * 2);
  s.tAdd = (_Float16*)ar.alloc(PN * CC * 2);
  s.tMh  = (_Float16*)ar.alloc(PN * CC * 2);
  s.tS   = (float*)ar.alloc((long)BB * HH * NN * TK * 4);
  s.tP   = (_Float16*)ar.alloc((long)BB * HH * NN * TK * 2);
  s.h1   = (float*)ar.alloc(PN * C2 * 4);
  s.h1h  = (_Float16*)ar.alloc(PN * C2 * 2);
  const long SK = (long)BB * TK;
  float* g1f = (float*)ar.alloc(SK*CC*4);   _Float16* g1h = (_Float16*)ar.alloc(SK*CC*2);
  float* g2f = (float*)ar.alloc(SK*CC*4);   _Float16* g2h = (_Float16*)ar.alloc(SK*CC*2);
  float* c1f = (float*)ar.alloc(SK*CC*4);   _Float16* c1h = (_Float16*)ar.alloc(SK*CC*2);
  float* c2f = (float*)ar.alloc(SK*CC*4);   _Float16* c2h = (_Float16*)ar.alloc(SK*CC*2);
  float* c1nf= (float*)ar.alloc(SK*CC*4);   _Float16* c1nh= (_Float16*)ar.alloc(SK*CC*2);
  float* c2nf= (float*)ar.alloc(SK*CC*4);   _Float16* c2nh= (_Float16*)ar.alloc(SK*CC*2);
  float* c1sf= (float*)ar.alloc(SK*CC*4);   _Float16* c1sh= (_Float16*)ar.alloc(SK*CC*2);
  float* c2sf= (float*)ar.alloc(SK*CC*4);   _Float16* c2sh= (_Float16*)ar.alloc(SK*CC*2);
  float* ccf = (float*)ar.alloc(SK*C2*4);   _Float16* cch = (_Float16*)ar.alloc(SK*C2*2);
  _Float16* confh1 = (_Float16*)ar.alloc(SK*C2*2);
  _Float16* confh2 = (_Float16*)ar.alloc(SK*C2*2);
  float* h2f = (float*)ar.alloc(SK*C2*4);
  float* sw  = (float*)ar.alloc(SK*4);
  float* uu  = (float*)ar.alloc((long)BB*DIM*4);
  float* vv  = (float*)ar.alloc((long)BB*DIM*4);
  float* Pout = (float*)d_out;

  // -------- stage 1: normalize descriptors, distance matrix, seeding
  k_l2norm<<<(int)PN, 128, 0, stream>>>(desc1, d1n, d1h);
  k_l2norm<<<(int)PN, 128, 0, stream>>>(desc2, d2n, d2h);
  { GemmP p = gp0();                               // dm = sqrt(|2 - 2 d1.d2^T|), f16
    p.A = d1h; p.sA1 = (long)NN*CC; p.lda = CC;
    p.Bm = d2h; p.sB1 = (long)NN*CC; p.ldb = CC;
    p.OH = dm; p.sO1 = (long)NN*NN; p.ldo = NN;
    p.M = NN; p.N = NN; p.K = CC;
    p.flags = GF_DMSQRT | GF_WH; gemm_launch(stream, p, BB); }
  k_row_top2<<<(int)PN, 256, 0, stream>>>(dm, v1, v2, nn1, NN);
  k_colmin<<<dim3(NN/256, BB), 256, 0, stream>>>(dm, nn2, NN);
  k_score<<<(int)(PN+255)/256, 256, 0, stream>>>(v1, v2, nn1, nn2, sc0, NN, (int)PN);
  k_gxy<<<(int)(PN+255)/256, 256, 0, stream>>>(x2, nn1, x2g, NN, (int)PN);
  k_fill<<<1, 64, 0, stream>>>(sums, 0.f, 8);
  k_pdist_sum<<<dim3(NN/256, BB), 256, 0, stream>>>(x1,  sums,     NN);
  k_pdist_sum<<<dim3(NN/256, BB), 256, 0, stream>>>(x2g, sums + 4, NN);
  k_localmax<<<dim3(NN, BB), 256, 0, stream>>>(sc0, sc1, x1, x2g, sums, sums + 4, NN);
  k_topk<<<BB, 256, 0, stream>>>(sc1, nn1, s1, s2, NN, TK);

  // -------- stage 2: positional encoding MLP -> a = pe(x) + d
  auto pe_apply = [&](const float* x, float* dnf, float* af, _Float16* ah) {
    k_pe0<<<(int)((PN*32+255)/256), 256, 0, stream>>>(x, pe.c0.W, pe.c0.b,
                                                      pe.bn0.g, pe.bn0.bt, t0h, (int)PN);
    { GemmP p = gp0();
      p.A = t0h; p.sA1 = (long)NN*32; p.lda = 32;
      p.Bm = pe.c1.Wh; p.ldb = 32;
      p.OH = t1h; p.sO1 = (long)NN*64; p.ldo = 64;
      p.bias = pe.c1.b; p.bnG = pe.bn1.g; p.bnB = pe.bn1.bt;
      p.M = NN; p.N = 64; p.K = 32;
      p.flags = GF_BIAS|GF_BN|GF_RELU|GF_WH; gemm_launch(stream, p, BB); }
    { GemmP p = gp0();
      p.A = t1h; p.sA1 = (long)NN*64; p.lda = 64;
      p.Bm = pe.c2.Wh; p.ldb = 64;
      p.OH = t2h; p.sO1 = (long)NN*128; p.ldo = 128;
      p.bias = pe.c2.b; p.bnG = pe.bn2.g; p.bnB = pe.bn2.bt;
      p.M = NN; p.N = 128; p.K = 64;
      p.flags = GF_BIAS|GF_BN|GF_RELU|GF_WH; gemm_launch(stream, p, BB); }
    { GemmP p = gp0();
      p.A = t2h; p.sA1 = (long)NN*128; p.lda = 128;
      p.Bm = pe.c3.Wh; p.ldb = 128;
      p.OH = t3h; p.sO1 = (long)NN*256; p.ldo = 256;
      p.bias = pe.c3.b; p.bnG = pe.bn3.g; p.bnB = pe.bn3.bt;
      p.M = NN; p.N = 256; p.K = 128;
      p.flags = GF_BIAS|GF_BN|GF_RELU|GF_WH; gemm_launch(stream, p, BB); }
    { GemmP p = gp0();
      p.A = t3h; p.sA1 = (long)NN*256; p.lda = 256;
      p.Bm = pe.c4.Wh; p.ldb = 256;
      p.OF = af; p.OH = ah; p.sO1 = (long)NN*CC; p.ldo = CC;
      p.bias = pe.c4.b;
      p.R = dnf; p.sR1 = (long)NN*CC; p.ldr = CC;
      p.M = NN; p.N = CC; p.K = 256;
      p.flags = GF_BIAS|GF_RESID|GF_WF|GF_WH; gemm_launch(stream, p, BB); }
  };
  pe_apply(x1, d1n, a1f, a1h);
  pe_apply(x2, d2n, a2f, a2h);

  // -------- stage 3: hybrid blocks
  Act a1{a1f, a1h, NN, CC}, a2{a2f, a2h, NN, CC};
  for (int bi = 0; bi < 6; ++bi) {
    HybW& w = blocks[bi];
    k_gather<<<dim3(TK, BB), 128, 0, stream>>>(a1f, a1h, s1, g1f, g1h, NN, TK);
    k_gather<<<dim3(TK, BB), 128, 0, stream>>>(a2f, a2h, s2, g2f, g2h, NN, TK);
    Act g1{g1f,g1h,TK,CC}, g2{g2f,g2h,TK,CC};
    Act c1{c1f,c1h,TK,CC}, c2{c2f,c2h,TK,CC};
    attn_apply(stream, w.down, g1, a1, nullptr, c1, s);
    attn_apply(stream, w.down, g2, a2, nullptr, c2, s);
    { GemmP p = gp0();                               // cf1 part A
      p.A = c1h; p.sA1 = (long)TK*CC; p.lda = CC;
      p.Bm = w.cf1.Wh; p.ldb = C2;
      p.OF = s.h1; p.sO1 = (long)TK*C2; p.ldo = C2;
      p.M = TK; p.N = C2; p.K = CC; p.flags = GF_WF; gemm_launch(stream, p, BB); }
    { GemmP p = gp0();                               // cf1 part B + bn/relu
      p.A = c2h; p.sA1 = (long)TK*CC; p.lda = CC;
      p.Bm = w.cf1.Wh + CC; p.ldb = C2;
      p.OF = s.h1; p.OH = s.h1h; p.sO1 = (long)TK*C2; p.ldo = C2;
      p.bias = w.cf1.b; p.bnG = w.cf_bn.g; p.bnB = w.cf_bn.bt;
      p.M = TK; p.N = C2; p.K = CC;
      p.flags = GF_ACCUM|GF_BIAS|GF_BN|GF_RELU|GF_WH; gemm_launch(stream, p, BB); }
    { GemmP p = gp0();                               // cf2
      p.A = s.h1h; p.sA1 = (long)TK*C2; p.lda = C2;
      p.Bm = w.cf2.Wh; p.ldb = C2;
      p.OF = ccf; p.OH = cch; p.sO1 = (long)TK*C2; p.ldo = C2;
      p.bias = w.cf2.b; p.M = TK; p.N = C2; p.K = C2;
      p.flags = GF_BIAS|GF_WF|GF_WH; gemm_launch(stream, p, BB); }
    Act ccA{ccf, cch, TK, C2}, ccB{ccf + CC, cch + CC, TK, C2};
    Act c1n{c1nf,c1nh,TK,CC}, c2n{c2nf,c2nh,TK,CC};
    Act c1s{c1sf,c1sh,TK,CC}, c2s{c2sf,c2sh,TK,CC};
    attn_apply(stream, w.cross, ccA, ccB, nullptr, c1n, s);
    attn_apply(stream, w.cross, ccB, ccA, nullptr, c2n, s);
    attn_apply(stream, w.self_, c1n, c1n, nullptr, c1s, s);
    attn_apply(stream, w.self_, c2n, c2n, nullptr, c2s, s);
    // confidence head
    k_inorm<<<dim3(CC, BB), 256, 0, stream>>>(c1sf, CC, TK, w.conf.bn1.g,
                                              w.conf.bn1.bt, confh1, C2);
    k_inorm<<<dim3(CC, BB), 256, 0, stream>>>(c2sf, CC, TK, w.conf.bn1.g + CC,
                                              w.conf.bn1.bt + CC, confh1 + CC, C2);
    { GemmP p = gp0();
      p.A = confh1; p.sA1 = (long)TK*C2; p.lda = C2;
      p.Bm = w.conf.c1.Wh; p.ldb = C2;
      p.OF = h2f; p.sO1 = (long)TK*C2; p.ldo = C2;
      p.bias = w.conf.c1.b; p.M = TK; p.N = C2; p.K = C2;
      p.flags = GF_BIAS|GF_WF; gemm_launch(stream, p, BB); }
    k_inorm<<<dim3(C2, BB), 256, 0, stream>>>(h2f, C2, TK, w.conf.bn2.g,
                                              w.conf.bn2.bt, confh2, C2);
    k_conf<<<(int)((SK+63)/64), 64, 0, stream>>>(confh2, w.conf.c2.W, w.conf.c2.b,
                                                 w.conf.sc.W, w.conf.sc.b,
                                                 c1sf, c2sf, sw, TK, (int)SK);
    attn_apply(stream, w.up, a1, c1s, sw, a1, s);
    attn_apply(stream, w.up, a2, c2s, sw, a2, s);
  }

  // -------- stage 4: final conv, matching matrix, sinkhorn
  { GemmP p = gp0();
    p.A = a1h; p.sA1 = (long)NN*CC; p.lda = CC;
    p.Bm = finalW.Wh; p.ldb = CC;
    p.OH = s.tQ; p.sO1 = (long)NN*CC; p.ldo = CC;
    p.bias = finalW.b; p.M = NN; p.N = CC; p.K = CC;
    p.flags = GF_BIAS|GF_WH; gemm_launch(stream, p, BB); }
  { GemmP p = gp0();
    p.A = a2h; p.sA1 = (long)NN*CC; p.lda = CC;
    p.Bm = finalW.Wh; p.ldb = CC;
    p.OH = s.tK; p.sO1 = (long)NN*CC; p.ldo = CC;
    p.bias = finalW.b; p.M = NN; p.N = CC; p.K = CC;
    p.flags = GF_BIAS|GF_WH; gemm_launch(stream, p, BB); }
  { GemmP p = gp0();                                 // M -> interior of d_out
    p.A = s.tQ; p.sA1 = (long)NN*CC; p.lda = CC;
    p.Bm = s.tK; p.sB1 = (long)NN*CC; p.ldb = CC;
    p.OF = Pout; p.sO1 = (long)DIM*DIM; p.ldo = DIM;
    p.M = NN; p.N = NN; p.K = CC;
    p.flags = GF_WF; gemm_launch(stream, p, BB); }
  k_border<<<dim3((DIM+255)/256, BB), 256, 0, stream>>>(Pout, dustbin, NN);
  k_softmax<<<dim3(DIM, BB), 256, 0, stream>>>(Pout, Pout, nullptr, nullptr,
                                               DIM, (long)DIM*DIM, 1, 0);
  k_fill<<<(int)((BB*DIM+255)/256), 256, 0, stream>>>(vv, 1.f, BB*DIM);
  for (int it = 0; it < 20; ++it) {                 // L2-resident (67MB < 192MB)
    k_sink_u<<<dim3(DIM, BB), 256, 0, stream>>>(Pout, vv, uu, NN);
    k_sink_v<<<dim3((DIM+255)/256, BB), 256, 0, stream>>>(Pout, uu, vv, NN);
  }
  k_sink_scale<<<dim3(DIM, BB), 256, 0, stream>>>(Pout, uu, vv, NN);
}
